// DenseConcatAttentionScore_56289841381548
// MI455X (gfx1250) — matrix operand order repaired and checked by reading, not compiled, not run on hardware
//
#include <hip/hip_runtime.h>
#include <hip/hip_bf16.h>

typedef _Float16 v4h  __attribute__((ext_vector_type(4)));
typedef _Float16 v8h  __attribute__((ext_vector_type(8)));
typedef _Float16 v16h __attribute__((ext_vector_type(16)));
typedef float    v8f  __attribute__((ext_vector_type(8)));

#define B_      32
#define T_      4096
#define D_      512
#define A_      512
#define MTILE   128
#define KSTRIDE 520   // padded LDS row stride (halves): 16 rows -> 16 distinct bank groups

// ---- CDNA5 async global->LDS copy (ASYNCcnt path) -------------------------
__device__ __forceinline__ uint32_t lds_off32(const void* p) {
  // generic pointers to LDS carry the byte offset in the low 32 bits
  return (uint32_t)(uintptr_t)p;
}
__device__ __forceinline__ void async_copy_b128(uint32_t lds_off, const void* g) {
  asm volatile("global_load_async_to_lds_b128 %0, %1, off"
               :: "v"(lds_off), "v"(g) : "memory");
}
#define WAIT_ASYNC(n) asm volatile("s_wait_asynccnt " #n ::: "memory")

// ---- branch-free tanh: 1 - 2/(exp(2x)+1)  (v_exp_f32 + v_rcp_f32) ---------
// Saturates correctly: x>>0 -> e=inf -> r=0 -> 1;  x<<0 -> e=0 -> r=1 -> -1.
__device__ __forceinline__ float fast_tanh(float x) {
  float e = __expf(2.0f * x);
  float r = __builtin_amdgcn_rcpf(e + 1.0f);
  return fmaf(-2.0f, r, 1.0f);
}

// ---------------------------------------------------------------------------
// Kernel 1: qpb[b][n] = sum_d q[b][d] * W[d][n] + bias[n]   (tiny)
// ---------------------------------------------------------------------------
__global__ __launch_bounds__(256) void prep_qpb_kernel(
    const float* __restrict__ q, const float* __restrict__ W,
    const float* __restrict__ bias, float* __restrict__ qpb) {
  int gid = blockIdx.x * 256 + threadIdx.x;   // 0..16383
  int bb  = gid >> 9;
  int n   = gid & (A_ - 1);
  float s = bias[n];
  const float* qrow = q + bb * D_;
#pragma unroll 8
  for (int d = 0; d < D_; ++d)
    s = fmaf(qrow[d], W[(size_t)d * A_ + n], s);
  qpb[gid] = s;
}

// ---------------------------------------------------------------------------
// Kernel 2: pack Wk = W[512:1024] (f32) into f16 WMMA-B fragment order.
// Tile t = nt*16+kc is one 32(K)x16(N) B fragment (512 halves = 1 KB):
//   region0 (halves   0..255): lane L halves h=0..7  at L*8+h
//   region1 (halves 256..511): lane L halves h=8..15 at 256+L*8+(h-8)
// Lane L (<16):  n = nt*16+L,    K = kc*32 + h (h<8), kc*32 + 8 + h (h>=8)
// Lane L (>=16): n = nt*16+L-16, K = kc*32 + 8 + h (h<8), kc*32 + 16 + h (h>=8)
// Split-region layout => each ds_load_b128 hits every bank-granule exactly
// twice (minimum 2 passes, conflict-free).
// ---------------------------------------------------------------------------
__global__ __launch_bounds__(512) void pack_wk_kernel(
    const float* __restrict__ W, _Float16* __restrict__ pack) {
  int t    = blockIdx.x;          // 0..511  (= nt*16 + kc)
  int nt   = t >> 4, kc = t & 15;
  int tid  = threadIdx.x;         // = part*256 + lane*8 + hh  -> coalesced write
  int part = tid >> 8;            // 0: halves 0-7, 1: halves 8-15
  int lane = (tid >> 3) & 31;
  int h    = part * 8 + (tid & 7);
  int n  = nt * 16 + (lane & 15);
  int kk = kc * 32 + part * 16 + ((lane >= 16) ? 8 : 0) + (tid & 7);
  pack[(size_t)t * 512 + tid] = (_Float16)W[(size_t)(D_ + kk) * A_ + n];
}

// ---------------------------------------------------------------------------
// Main kernel: one 128-row t-tile of one batch per WG.
//  - k-tile staged once f32->f16 in LDS (k read exactly once from HBM)
//  - Wk streamed in 8 KB half-panels via async global->LDS, double buffered
//  - 8x v_wmma_f32_16x16x32_f16 per half-panel, fused tanh+dot epilogue
// ---------------------------------------------------------------------------
__global__ __launch_bounds__(256) void score_kernel(
    const float* __restrict__ kg, const _Float16* __restrict__ wkpack,
    const float* __restrict__ qpb, const float* __restrict__ vvec,
    float* __restrict__ out) {
  __shared__ __align__(16) _Float16 kbuf[MTILE * KSTRIDE];   // 133,120 B
  __shared__ __align__(16) _Float16 wbuf[2][8 * 512];        // 2 x 8,192 B

  const int tid  = threadIdx.x;
  const int lane = tid & 31;
  const int wv   = tid >> 5;                   // wave 0..7 owns rows wv*16..+15
  const int bb   = blockIdx.y;
  const int t0   = blockIdx.x * MTILE;

  // ---- stage k tile: 128 rows x 512 f32 -> f16 in LDS (coalesced float4) ----
  const float4* ksrc = (const float4*)(kg + ((size_t)bb * T_ + t0) * D_);
#pragma unroll 8
  for (int i = 0; i < 64; ++i) {
    int c   = i * 256 + tid;                   // float4 index 0..16383
    int row = c >> 7, c4 = c & 127;
    float4 f = ksrc[row * 128 + c4];
    v4h hh = { (_Float16)f.x, (_Float16)f.y, (_Float16)f.z, (_Float16)f.w };
    *(v4h*)(kbuf + row * KSTRIDE + c4 * 4) = hh;
  }

  const int laneN = lane & 15;
  const int rw    = wv * 16;
  const _Float16* abase = kbuf + (rw + laneN) * KSTRIDE + ((lane >> 4) << 3);

  float score8[8];
#pragma unroll
  for (int i = 0; i < 8; ++i) score8[i] = 0.f;

  const char* wsrc = (const char*)wkpack;

  // ---- prefetch half-panel 0 (async, no VGPR round-trip) ----
#pragma unroll
  for (int j = 0; j < 2; ++j) {
    int idx = j * 256 + tid;                   // uint4 chunks of the 8 KB panel
    async_copy_b128(lds_off32(&wbuf[0][0]) + idx * 16, wsrc + idx * 16);
  }

  // half-panel p = nt*2 + half covers kc = half*8 .. half*8+7 of N-tile nt
  for (int nt = 0; nt < 32; ++nt) {
    // epilogue operands: issue early so latency hides under the WMMA stream
    int   n  = nt * 16 + laneN;
    float qn = qpb[bb * A_ + n];
    float vn = vvec[n];

    v8f acc = {};
#pragma unroll
    for (int half = 0; half < 2; ++half) {
      int p = nt * 2 + half;
      if (p < 63) {                            // prefetch next half-panel
        const char* g = wsrc + (size_t)(p + 1) * 8192;
        uint32_t l = lds_off32(&wbuf[half ^ 1][0]);
#pragma unroll
        for (int j = 0; j < 2; ++j) {
          int idx = j * 256 + tid;
          async_copy_b128(l + idx * 16, g + idx * 16);
        }
        WAIT_ASYNC(2);                         // panel p landed; p+1 in flight
      } else {
        WAIT_ASYNC(0);
      }
      __syncthreads();                         // panel p visible to all waves

      const _Float16* wb = &wbuf[half][0];
      const _Float16* ab = abase + half * 256; // kc base 0 or 8 (x32 halves)
#pragma unroll
      for (int kc = 0; kc < 8; ++kc) {
        v8h alo = *(const v8h*)(ab + kc * 32);        // K 0..7   (hi lanes: 8..15)
        v8h ahi = *(const v8h*)(ab + kc * 32 + 16);   // K 16..23 (hi lanes: 24..31)
        v16h afrag = __builtin_shufflevector(alo, ahi,
            0,1,2,3,4,5,6,7,8,9,10,11,12,13,14,15);
        v8h blo = *(const v8h*)(wb + kc * 512 + lane * 8);        // halves 0-7
        v8h bhi = *(const v8h*)(wb + kc * 512 + 256 + lane * 8);  // halves 8-15
        v16h bfrag = __builtin_shufflevector(blo, bhi,
            0,1,2,3,4,5,6,7,8,9,10,11,12,13,14,15);
        acc = __builtin_amdgcn_wmma_f32_16x16x32_f16(
            /*neg_a=*/false, afrag, /*neg_b=*/false, bfrag,
            /*c_mod=*/(short)0, acc, /*reuse_a=*/false, /*reuse_b=*/false);
      }
      __syncthreads();                         // all done reading wbuf[half]
    }

    // ---- fused epilogue: tanh(kp + qp + bias) * v, accumulate per row ----
#pragma unroll
    for (int i = 0; i < 8; ++i)
      score8[i] += fast_tanh(acc[i] + qn) * vn;  // C layout: vgpr i -> M=i (+8 hi lanes)
  }

  // ---- reduce the 16 lanes (N values) sharing each row ----
#pragma unroll
  for (int m = 8; m >= 1; m >>= 1)
#pragma unroll
    for (int i = 0; i < 8; ++i)
      score8[i] += __shfl_xor(score8[i], m, 32);

  if (laneN == 0) {                            // lanes 0 (rows +0..7) and 16 (+8..15)
    float* op = out + (size_t)bb * T_ + t0 + rw + ((lane >> 4) << 3);
#pragma unroll
    for (int i = 0; i < 8; ++i) op[i] = score8[i];
  }
}

// ---------------------------------------------------------------------------
extern "C" void kernel_launch(void* const* d_in, const int* in_sizes, int n_in,
                              void* d_out, int out_size, void* d_ws, size_t ws_size,
                              hipStream_t stream) {
  const float* q  = (const float*)d_in[0];   // (32, 512)
  const float* k  = (const float*)d_in[1];   // (32, 4096, 512)
  const float* W  = (const float*)d_in[2];   // (1024, 512)
  const float* bs = (const float*)d_in[3];   // (512,)
  const float* v  = (const float*)d_in[4];   // (512,)
  float* out = (float*)d_out;                // (32, 4096)

  float*    qpb  = (float*)d_ws;                         // 64 KB
  _Float16* pack = (_Float16*)((char*)d_ws + 65536);     // 512 KB

  prep_qpb_kernel<<<64, 256, 0, stream>>>(q, W, bs, qpb);
  pack_wk_kernel<<<512, 512, 0, stream>>>(W, pack);
  score_kernel<<<dim3(T_ / MTILE, B_), 256, 0, stream>>>(k, pack, qpb, v, out);
}
